// DecoderLayer_8985071583694
// MI455X (gfx1250) — matrix operand order repaired and checked by reading, not compiled, not run on hardware
//
#include <hip/hip_runtime.h>
#include <hip/hip_bf16.h>

// Problem constants (match reference)
#define BB  4
#define LL  1024
#define SS  1024
#define DD  1024
#define HH  16
#define HE  64      // head dim = D/H
#define DFFC 4096

typedef __attribute__((ext_vector_type(16))) _Float16 v16h;
typedef __attribute__((ext_vector_type(8)))  _Float16 v8h;
typedef __attribute__((ext_vector_type(8)))  float    v8f;

static __device__ __forceinline__ v8f wmma_f16(v16h a, v16h b, v8f c) {
  // D = A(16x32 f16) * B(32x16 f16) + C(16x16 f32)
  return __builtin_amdgcn_wmma_f32_16x16x32_f16(
      /*neg_a=*/false, a, /*neg_b=*/false, b,
      /*c_mod=*/(short)0, c, /*reuse_a=*/false, /*reuse_b=*/false);
}

static __device__ __forceinline__ v16h cat8(v8h lo, v8h hi) {
  return __builtin_shufflevector(lo, hi, 0, 1, 2, 3, 4, 5, 6, 7,
                                 8, 9, 10, 11, 12, 13, 14, 15);
}

// ---- fragment loaders (k = 16*half + j in A and B alike) ----
// A 16x32 (MxK), row-major source. Element j: k = 16*half + j
// => one contiguous 16-half run per lane -> two b128 loads.
static __device__ __forceinline__ v16h load_A16x32(
    const _Float16* __restrict__ src, int lda, int lane) {
  const int m = lane & 15, h = lane >> 4;
  const _Float16* row = src + (size_t)m * lda + 16 * h;
  return cat8(*(const v8h*)(row), *(const v8h*)(row + 8));
}

// B 32x16 (KxN) held as N x K row-major (B[k][n] = W[n][k]).
// Element j: k = 16*half + j -> one contiguous 32B run per lane.
static __device__ __forceinline__ v16h load_B32x16_colmajor(
    const _Float16* __restrict__ w, int ldw, int lane) {
  const int n = lane & 15, h = lane >> 4;
  return *(const v16h*)(w + (size_t)n * ldw + 16 * h);
}

// Same as above but 16B-granular (for padded LDS tiles).
static __device__ __forceinline__ v16h load_B_lds(
    const _Float16* w, int ldw, int lane) {
  const int n = lane & 15, h = lane >> 4;
  const _Float16* p = w + n * ldw + 16 * h;
  return cat8(*(const v8h*)(p), *(const v8h*)(p + 8));
}

// gfx1250 async copy: global -> LDS (16B per lane), tracked by ASYNCcnt.
static __device__ __forceinline__ void async_b128_to_lds(
    unsigned lds_off, const void* gptr) {
  asm volatile("global_load_async_to_lds_b128 %0, %1, off"
               :: "v"(lds_off), "v"((unsigned long long)gptr) : "memory");
}
static __device__ __forceinline__ void wait_asynccnt0() {
  asm volatile("s_wait_asynccnt 0x0" ::: "memory");
}

// ---------------------------------------------------------------------------
// fp32 -> f16 conversion
__global__ void __launch_bounds__(256) cast_kernel(
    const float* __restrict__ in, _Float16* __restrict__ out, int n) {
  int i = blockIdx.x * blockDim.x + threadIdx.x;
  const int stride = gridDim.x * blockDim.x;
  for (; i < n; i += stride) out[i] = (_Float16)in[i];
}

// ---------------------------------------------------------------------------
// Per-batch transpose + cast: in[b*rows + s][d] (f32) -> outT[b*cols + d][s] (f16)
__global__ void __launch_bounds__(256) transpose_cast_kernel(
    const float* __restrict__ in, _Float16* __restrict__ outT,
    int rows, int cols) {
  __shared__ float tile[32][33];
  const int b  = blockIdx.z;
  const int d0 = blockIdx.x * 32;
  const int s0 = blockIdx.y * 32;
  const int tx = threadIdx.x & 31;
  const int ty = threadIdx.x >> 5;
#pragma unroll
  for (int i = ty; i < 32; i += 8)
    tile[i][tx] = in[((size_t)b * rows + s0 + i) * cols + d0 + tx];
  __syncthreads();
#pragma unroll
  for (int i = ty; i < 32; i += 8)
    outT[((size_t)b * cols + d0 + i) * rows + s0 + tx] = (_Float16)tile[tx][i];
}

// ---------------------------------------------------------------------------
// Flash-attention: one wave per (b, h, 16-row tile of L). f16 WMMA, fp32 softmax.
// K is [b][s][d] (contiguous in e); Vt is [b][d][s] (contiguous in s).
__global__ void __launch_bounds__(256) attn_kernel(
    const _Float16* __restrict__ Q, const _Float16* __restrict__ Km,
    const _Float16* __restrict__ Vt, float* __restrict__ Out,
    int Lq, int Skv, int causal) {
  __shared__ _Float16 pst[8][16 * 32];     // per-wave P staging (1KB each)
  const int lane  = threadIdx.x & 31;
  const int wv    = threadIdx.x >> 5;
  const int ltiles = Lq >> 4;
  const int wid = blockIdx.x * 8 + wv;
  if (wid >= BB * HH * ltiles) return;     // wave-uniform (grid exact anyway)
  const int lt = wid % ltiles;
  const int h  = (wid / ltiles) % HH;
  const int b  = wid / (ltiles * HH);
  const int l0 = lt << 4;
  const int m     = lane & 15;
  const int half  = lane >> 4;
  const int mbase = half * 8;

  const _Float16* qbase = Q + ((size_t)b * Lq + l0) * DD + h * HE;
  const v16h aq0 = load_A16x32(qbase, DD, lane);        // e = 0..31
  const v16h aq1 = load_A16x32(qbase + 32, DD, lane);   // e = 32..63

  v8f o0 = {}, o1 = {}, o2 = {}, o3 = {};
  float m8[8], l8[8];
#pragma unroll
  for (int r = 0; r < 8; ++r) { m8[r] = -3.0e38f; l8[r] = 0.f; }

  const int sEnd = causal ? (l0 + 16) : Skv;
  for (int s0 = 0; s0 < sEnd; s0 += 32) {
    // ---- load 4 K fragments up front (compiler clauses + partial waits) ----
    const _Float16* kb0 = Km + ((size_t)b * Skv + s0) * DD + h * HE;
    const _Float16* kb1 = kb0 + (size_t)16 * DD;
    const v16h bk00 = load_B32x16_colmajor(kb0,      DD, lane);
    const v16h bk01 = load_B32x16_colmajor(kb0 + 32, DD, lane);
    const v16h bk10 = load_B32x16_colmajor(kb1,      DD, lane);
    const v16h bk11 = load_B32x16_colmajor(kb1 + 32, DD, lane);
    v8f c0 = {}, c1 = {};
    c0 = wmma_f16(aq0, bk00, c0);
    c0 = wmma_f16(aq1, bk01, c0);
    c1 = wmma_f16(aq0, bk10, c1);
    c1 = wmma_f16(aq1, bk11, c1);

    // ---- issue V fragment loads now; latency hides under softmax ----
    const _Float16* vb = Vt + ((size_t)b * DD + h * HE) * Skv + s0;
    const v16h bv0 = load_B32x16_colmajor(vb,                    Skv, lane);
    const v16h bv1 = load_B32x16_colmajor(vb + (size_t)16 * Skv, Skv, lane);
    const v16h bv2 = load_B32x16_colmajor(vb + (size_t)32 * Skv, Skv, lane);
    const v16h bv3 = load_B32x16_colmajor(vb + (size_t)48 * Skv, Skv, lane);

    const float scale = 0.125f;            // 1/sqrt(64)
#pragma unroll
    for (int r = 0; r < 8; ++r) { c0[r] *= scale; c1[r] *= scale; }

    if (causal && (s0 + 31 > l0)) {
#pragma unroll
      for (int r = 0; r < 8; ++r) {
        const int lrow = l0 + mbase + r;
        if (s0 + m      > lrow) c0[r] = -3.0e38f;
        if (s0 + 16 + m > lrow) c1[r] = -3.0e38f;
      }
    }

    // ---- online softmax (rows in 16-lane halves; reduce over n) ----
    float nm[8];
#pragma unroll
    for (int r = 0; r < 8; ++r) {
      float t = fmaxf(c0[r], c1[r]);
#pragma unroll
      for (int off = 8; off > 0; off >>= 1)
        t = fmaxf(t, __shfl_xor(t, off, 32));
      nm[r] = fmaxf(m8[r], t);
    }
    float p0[8], p1[8];
#pragma unroll
    for (int r = 0; r < 8; ++r) {
      const float alpha = __expf(m8[r] - nm[r]);
      p0[r] = __expf(c0[r] - nm[r]);
      p1[r] = __expf(c1[r] - nm[r]);
      float rs = p0[r] + p1[r];
#pragma unroll
      for (int off = 8; off > 0; off >>= 1)
        rs += __shfl_xor(rs, off, 32);
      l8[r] = l8[r] * alpha + rs;
      m8[r] = nm[r];
      o0[r] *= alpha; o1[r] *= alpha; o2[r] *= alpha; o3[r] *= alpha;
    }

    // ---- relayout P (C-frag) -> A-frag via per-wave LDS tile ----
#pragma unroll
    for (int r = 0; r < 8; ++r) {
      pst[wv][(mbase + r) * 32 + m]      = (_Float16)p0[r];
      pst[wv][(mbase + r) * 32 + 16 + m] = (_Float16)p1[r];
    }
    asm volatile("" ::: "memory");         // same-wave DS ops are in order
    const _Float16* prow = &pst[wv][m * 32 + 16 * half];
    const v16h ap = cat8(*(const v8h*)(prow), *(const v8h*)(prow + 8));

    // ---- O += P(16x32) * V(32x64) ----
    o0 = wmma_f16(ap, bv0, o0);
    o1 = wmma_f16(ap, bv1, o1);
    o2 = wmma_f16(ap, bv2, o2);
    o3 = wmma_f16(ap, bv3, o3);
  }

  float* ob = Out + ((size_t)b * Lq + l0) * DD + h * HE;
#pragma unroll
  for (int r = 0; r < 8; ++r) {
    const float inv = 1.0f / l8[r];
    float* row = ob + (size_t)(mbase + r) * DD;
    row[ 0 + m] = o0[r] * inv;
    row[16 + m] = o1[r] * inv;
    row[32 + m] = o2[r] * inv;
    row[48 + m] = o3[r] * inv;
  }
}

// ---------------------------------------------------------------------------
// residual + LayerNorm over D=1024; one block (8 waves) per row.
__global__ void __launch_bounds__(256) res_ln_kernel(
    const float* __restrict__ A, const float* __restrict__ Bq,
    const float* __restrict__ g, const float* __restrict__ be,
    float* __restrict__ outf, _Float16* __restrict__ outh) {
  __shared__ float rs1[8], rs2[8];
  const size_t base = (size_t)blockIdx.x * DD;
  const int tid = threadIdx.x;
  float s1 = 0.f, s2 = 0.f, xv[4];
#pragma unroll
  for (int j = 0; j < 4; ++j) {
    const int i = tid + j * 256;
    const float x = A[base + i] + Bq[base + i];
    xv[j] = x; s1 += x; s2 += x * x;
  }
#pragma unroll
  for (int off = 16; off > 0; off >>= 1) {
    s1 += __shfl_xor(s1, off, 32);
    s2 += __shfl_xor(s2, off, 32);
  }
  if ((tid & 31) == 0) { rs1[tid >> 5] = s1; rs2[tid >> 5] = s2; }
  __syncthreads();
  float t1 = 0.f, t2 = 0.f;
#pragma unroll
  for (int w = 0; w < 8; ++w) { t1 += rs1[w]; t2 += rs2[w]; }
  const float mu   = t1 * (1.0f / DD);
  const float var  = t2 * (1.0f / DD) - mu * mu;
  const float rstd = rsqrtf(var + 1e-5f);
#pragma unroll
  for (int j = 0; j < 4; ++j) {
    const int i = tid + j * 256;
    const float y = (xv[j] - mu) * rstd * g[i] + be[i];
    outf[base + i] = y;
    if (outh) outh[base + i] = (_Float16)y;
  }
}

// ---------------------------------------------------------------------------
// Block-cooperative GEMM: out = act(A[MxK] * W[NxK]^T + bias).
// 256-thread block computes 128x128; the shared 128x32 B panel is streamed
// into double-buffered LDS with gfx1250 async loads (ASYNCcnt), so the 8
// waves read B from LDS instead of 8x redundant L2 traffic.
#define BPAD 40   // LDS row pitch in halves (32 data + 8 pad -> bank spread)
__global__ void __launch_bounds__(256) gemm_kernel(
    const _Float16* __restrict__ A, const _Float16* __restrict__ W,
    const float* __restrict__ bias, _Float16* __restrict__ outh,
    float* __restrict__ outf, int M, int N, int K, int relu) {
  __shared__ __align__(16) _Float16 tileB[2][128 * BPAD];
  const int tid  = threadIdx.x;
  const int lane = tid & 31;
  const int wv   = tid >> 5;
  const int tilesN = N >> 7;
  const int m0 = (blockIdx.x / tilesN) * 128 + wv * 16;
  const int n0 = (blockIdx.x % tilesN) * 128;
  const int mbase = (lane >> 4) * 8;
  const int n = lane & 15;

  // B panel staging assignment: thread -> (row, 32B run)
  const int srow = tid >> 1;           // 0..127
  const int skof = (tid & 1) * 16;     // halves 0 or 16
  const _Float16* wbase = W + (size_t)n0 * K;
  const _Float16* gsrc = wbase + (size_t)srow * K + skof;

  // prologue: stage k0 = 0 panel into buffer 0
  {
    unsigned l0 = (unsigned)(uintptr_t)&tileB[0][srow * BPAD + skof];
    async_b128_to_lds(l0, gsrc);
    async_b128_to_lds(l0 + 16, gsrc + 8);
  }

  v8f c[8] = {};
  const _Float16* arow = A + (size_t)m0 * K;
  for (int k0 = 0; k0 < K; k0 += 32) {
    const int buf = (k0 >> 5) & 1;
    wait_asynccnt0();                  // our panel chunks have landed in LDS
    __syncthreads();                   // everyone's chunks visible, prev reads done
    if (k0 + 32 < K) {                 // stream next panel into other buffer
      unsigned ld = (unsigned)(uintptr_t)&tileB[buf ^ 1][srow * BPAD + skof];
      const _Float16* gs = gsrc + k0 + 32;
      async_b128_to_lds(ld, gs);
      async_b128_to_lds(ld + 16, gs + 8);
    }
    __builtin_prefetch(arow + k0 + 128, 0, 1);
    const v16h a = load_A16x32(arow + k0, K, lane);
#pragma unroll
    for (int t = 0; t < 8; ++t)
      c[t] = wmma_f16(a, load_B_lds(&tileB[buf][(t * 16) * BPAD], BPAD, lane),
                      c[t]);
  }
#pragma unroll
  for (int t = 0; t < 8; ++t) {
    const int col = n0 + t * 16 + n;
    const float bv = bias ? bias[col] : 0.f;
#pragma unroll
    for (int r = 0; r < 8; ++r) {
      float val = c[t][r] + bv;
      if (relu) val = fmaxf(val, 0.f);
      const size_t idx = (size_t)(m0 + mbase + r) * N + col;
      if (outh) outh[idx] = (_Float16)val;
      if (outf) outf[idx] = val;
    }
  }
}

// ---------------------------------------------------------------------------
extern "C" void kernel_launch(void* const* d_in, const int* in_sizes, int n_in,
                              void* d_out, int out_size, void* d_ws, size_t ws_size,
                              hipStream_t stream) {
  (void)in_sizes; (void)n_in; (void)out_size; (void)ws_size;
  const float* x     = (const float*)d_in[0];
  const float* cross = (const float*)d_in[1];
  // d_in[2] = x_mask (causal, handled analytically)
  const float* w1  = (const float*)d_in[3];
  const float* b1  = (const float*)d_in[4];
  const float* w2  = (const float*)d_in[5];
  const float* b2  = (const float*)d_in[6];
  const float* g1  = (const float*)d_in[7];
  const float* be1 = (const float*)d_in[8];
  const float* g2  = (const float*)d_in[9];
  const float* be2 = (const float*)d_in[10];
  const float* g3  = (const float*)d_in[11];
  const float* be3 = (const float*)d_in[12];
  float* out = (float*)d_out;

  char* ws = (char*)d_ws;
  size_t off = 0;
  auto alloc = [&](size_t bytes) -> char* {
    char* p = ws + off; off += (bytes + 255) & ~(size_t)255; return p;
  };
  const size_t nXD = (size_t)BB * LL * DD;
  _Float16* xh  = (_Float16*)alloc(nXD * 2);                 // x / LN1 out (f16)
  _Float16* ch  = (_Float16*)alloc(nXD * 2);                 // cross / LN2 out (f16)
  _Float16* vtx = (_Float16*)alloc(nXD * 2);                 // x transposed [b][d][s]
  _Float16* vtc = (_Float16*)alloc(nXD * 2);                 // cross transposed
  _Float16* w1h = (_Float16*)alloc((size_t)DFFC * DD * 2);
  _Float16* w2h = (_Float16*)alloc((size_t)DD * DFFC * 2);
  _Float16* yh  = (_Float16*)alloc((size_t)BB * LL * DFFC * 2);
  float* af = (float*)alloc(nXD * 4);                        // attn / FFN2 out
  float* f1 = (float*)alloc(nXD * 4);                        // after LN1 (f32)
  float* f2 = (float*)alloc(nXD * 4);                        // after LN2 (f32)

  cast_kernel<<<1024, 256, 0, stream>>>(x,     xh,  (int)nXD);
  cast_kernel<<<1024, 256, 0, stream>>>(cross, ch,  (int)nXD);
  cast_kernel<<<1024, 256, 0, stream>>>(w1,    w1h, DFFC * DD);
  cast_kernel<<<1024, 256, 0, stream>>>(w2,    w2h, DD * DFFC);
  transpose_cast_kernel<<<dim3(32, 32, BB), 256, 0, stream>>>(x,     vtx, LL, DD);
  transpose_cast_kernel<<<dim3(32, 32, BB), 256, 0, stream>>>(cross, vtc, SS, DD);

  // self-attention (causal); waves = 4*16*64 = 4096 -> 512 blocks of 8 waves
  attn_kernel<<<512, 256, 0, stream>>>(xh, xh, vtx, af, LL, LL, 1);
  res_ln_kernel<<<BB * LL, 256, 0, stream>>>(x, af, g1, be1, f1, xh);
  // cross-attention (no mask)
  attn_kernel<<<512, 256, 0, stream>>>(xh, ch, vtc, af, LL, SS, 0);
  res_ln_kernel<<<BB * LL, 256, 0, stream>>>(f1, af, g2, be2, f2, ch);
  // FFN: block tiles 128x128; grids exact
  gemm_kernel<<<1024, 256, 0, stream>>>(ch, w1h, b1, yh, nullptr,
                                        BB * LL, DFFC, DD, 1);
  gemm_kernel<<<256, 256, 0, stream>>>(yh, w2h, b2, nullptr, af,
                                       BB * LL, DD, DFFC, 0);
  res_ln_kernel<<<BB * LL, 256, 0, stream>>>(f2, af, g3, be3, out, nullptr);
}
